// ChenDifferentiableAllocator_26268019982504
// MI455X (gfx1250) — compile-verified, matrix operand order checked by reading, not run on hardware
//
#include <hip/hip_runtime.h>
#include <hip/hip_bf16.h>

// DynOT Sinkhorn (L=64, B=6, 200 log-domain iterations) as a single persistent
// wave32 kernel for MI455X / gfx1250.
//
// Latency-bound problem: all state lives in VGPRs of one wave.
//  - lane l owns rows l and l+32 of the 64x6 kernel matrix K
//  - column reduction (over 64 rows) uses:
//      * 5-step shfl_xor butterfly for the per-column max (stabilization)
//      * two chained v_wmma_f32_16x16x32_f16 (B = ones) for the per-column
//        exp-sum: D[m][n] = sum_k A[m][k]  -> 64-row reduction in 2 matrix ops
//    exp(t - max) is in [0,1], so the f16 A-operand is numerically safe;
//    accumulation is f32.

typedef __attribute__((ext_vector_type(16))) _Float16 v16h;
typedef __attribute__((ext_vector_type(8)))  float    v8f;

#define LROWS 64
#define NB    6
#define EPS   0.02f
#define NITER 200

__global__ __launch_bounds__(32)
void dynot_sinkhorn_gfx1250(const float* __restrict__ trH,
                            const float* __restrict__ wmax,
                            const float* __restrict__ avec,
                            const float* __restrict__ theta,
                            const float* __restrict__ phi,
                            const int*   __restrict__ bits,
                            float* __restrict__ out)
{
    // Staging buffers for the two WMMA A-operands (rows 0..31 and 32..63),
    // stored directly in the 16-bit A 16x32 VGPR layout so the consumer side
    // is two contiguous ds_load_b128 per lane.
    __shared__ __align__(32) _Float16 ldsA[2][32][16];

    const int lane = threadIdx.x;          // 0..31, one wave32
    const int r0 = lane, r1 = lane + 32;   // rows owned by this lane

    // Zero LDS once: A rows m=6..15 are never written and must contribute 0.
    {
        _Float16* p = &ldsA[0][0][0];
        #pragma unroll
        for (int i = 0; i < 32; ++i) p[lane + 32 * i] = (_Float16)0.0f;
    }
    __syncthreads();

    // ---- uniform per-column data -------------------------------------------
    float lb[NB];     // log(softmax(phi))
    {
        float ph[NB];
        #pragma unroll
        for (int j = 0; j < NB; ++j) ph[j] = phi[j];
        float pm = ph[0];
        #pragma unroll
        for (int j = 1; j < NB; ++j) pm = fmaxf(pm, ph[j]);
        float ps = 0.0f;
        #pragma unroll
        for (int j = 0; j < NB; ++j) ps += __expf(ph[j] - pm);
        const float lse = pm + __logf(ps);
        #pragma unroll
        for (int j = 0; j < NB; ++j) lb[j] = ph[j] - lse;
    }

    // ---- per-row data: K = (theta - C)/eps, log_a --------------------------
    float K0[NB], K1[NB];
    const float tH0 = trH[r0],  tH1 = trH[r1];
    const float w0  = wmax[r0], w1  = wmax[r1];
    const float la0 = __logf(avec[r0] + 1e-40f);
    const float la1 = __logf(avec[r1] + 1e-40f);
    #pragma unroll
    for (int j = 0; j < NB; ++j) {
        const float dn = exp2f((float)bits[j]) - 1.0f;
        const float d0 = 2.0f * w0 / dn;
        const float d1 = 2.0f * w1 / dn;
        const float C0 = 0.5f * tH0 * (d0 * d0 * (1.0f / 12.0f));
        const float C1 = 0.5f * tH1 * (d1 * d1 * (1.0f / 12.0f));
        K0[j] = (theta[r0 * NB + j] - C0) * (1.0f / EPS);
        K1[j] = (theta[r1 * NB + j] - C1) * (1.0f / EPS);
    }

    // WMMA A-layout scatter coordinates for this lane (k == lane for both
    // A-matrices: A1 covers rows 0..31 (row r0), A2 rows 32..63 (row r1)).
    //   element (m,k):  lane = m + 16*((k>>3)&1),  vgpr = ((k&7)>>1) + 4*(k>>4),
    //                   halfword = k&1
    const int halfSel = (lane >> 3) & 1;
    const int slot    = ((((lane & 7) >> 1) + 4 * (lane >> 4)) << 1) | (lane & 1);

    // B = ones(32x16) in f16 (every element 1.0 regardless of layout).
    v16h bones;
    #pragma unroll
    for (int i = 0; i < 16; ++i) bones[i] = (_Float16)1.0f;

    float f0 = 0.0f, f1 = 0.0f;
    float g[NB];
    #pragma unroll
    for (int j = 0; j < NB; ++j) g[j] = 0.0f;

    // ---- 200 alternating Sinkhorn updates ----------------------------------
    for (int it = 0; it < NITER; ++it) {
        // f-update: pure per-lane VALU (row LSE over 6 columns)
        {
            float t0[NB], t1[NB];
            float m0 = -3.0e38f, m1 = -3.0e38f;
            #pragma unroll
            for (int j = 0; j < NB; ++j) {
                t0[j] = K0[j] + g[j];  m0 = fmaxf(m0, t0[j]);
                t1[j] = K1[j] + g[j];  m1 = fmaxf(m1, t1[j]);
            }
            float s0 = 0.0f, s1 = 0.0f;
            #pragma unroll
            for (int j = 0; j < NB; ++j) {
                s0 += __expf(t0[j] - m0);
                s1 += __expf(t1[j] - m1);
            }
            f0 = la0 - (m0 + __logf(s0));
            f1 = la1 - (m1 + __logf(s1));
        }

        // g-update: column LSE over 64 rows
        float u0[NB], u1[NB], mj[NB];
        #pragma unroll
        for (int j = 0; j < NB; ++j) {
            u0[j] = K0[j] + f0;
            u1[j] = K1[j] + f1;
            mj[j] = fmaxf(u0[j], u1[j]);
        }
        // per-column max: 5-step wave32 butterfly (ds_swizzle / permlane)
        #pragma unroll
        for (int d = 16; d >= 1; d >>= 1) {
            #pragma unroll
            for (int j = 0; j < NB; ++j)
                mj[j] = fmaxf(mj[j], __shfl_xor(mj[j], d, 32));
        }
        // stage exp(u - m) into the WMMA A layout (values in [0,1] -> f16 safe)
        #pragma unroll
        for (int j = 0; j < NB; ++j) {
            ldsA[0][j + 16 * halfSel][slot] = (_Float16)__expf(u0[j] - mj[j]);
            ldsA[1][j + 16 * halfSel][slot] = (_Float16)__expf(u1[j] - mj[j]);
        }
        __syncthreads();                       // DS store -> load ordering
        v16h A1 = *(const v16h*)&ldsA[0][lane][0];
        v16h A2 = *(const v16h*)&ldsA[1][lane][0];
        __syncthreads();                       // loads done before next stores

        // column sums: D[m][n] = sum_k A[m][k] * 1, chained over both halves
        v8f acc = {};
        acc = __builtin_amdgcn_wmma_f32_16x16x32_f16(
                  false, A1, false, bones, (short)0, acc, false, false);
        acc = __builtin_amdgcn_wmma_f32_16x16x32_f16(
                  false, A2, false, bones, (short)0, acc, false, false);

        // s_j sits in D VGPR j at lanes 0..15 (n = 0..15); broadcast from lane 0
        #pragma unroll
        for (int j = 0; j < NB; ++j) {
            const float sj = __uint_as_float(
                __builtin_amdgcn_readfirstlane(__float_as_uint(acc[j])));
            g[j] = lb[j] - (mj[j] + __logf(sj));
        }
    }

    // ---- P = exp(K + f + g), normalized by global sum ----------------------
    float p0[NB], p1[NB];
    float part = 0.0f;
    #pragma unroll
    for (int j = 0; j < NB; ++j) {
        p0[j] = __expf(K0[j] + f0 + g[j]);
        p1[j] = __expf(K1[j] + f1 + g[j]);
        part += p0[j] + p1[j];
    }
    #pragma unroll
    for (int d = 16; d >= 1; d >>= 1) part += __shfl_xor(part, d, 32);
    const float inv = 1.0f / (part + 1e-40f);
    #pragma unroll
    for (int j = 0; j < NB; ++j) {
        out[r0 * NB + j] = p0[j] * inv;
        out[r1 * NB + j] = p1[j] * inv;
    }
}

extern "C" void kernel_launch(void* const* d_in, const int* in_sizes, int n_in,
                              void* d_out, int out_size, void* d_ws, size_t ws_size,
                              hipStream_t stream) {
    (void)in_sizes; (void)n_in; (void)out_size; (void)d_ws; (void)ws_size;
    const float* trH   = (const float*)d_in[0];
    const float* wmax  = (const float*)d_in[1];
    const float* avec  = (const float*)d_in[2];
    const float* theta = (const float*)d_in[3];
    const float* phi   = (const float*)d_in[4];
    const int*   bits  = (const int*)d_in[5];
    float*       out   = (float*)d_out;

    // Entire 200-iteration Sinkhorn in one persistent wave32 launch.
    dynot_sinkhorn_gfx1250<<<dim3(1), dim3(32), 0, stream>>>(
        trH, wmax, avec, theta, phi, bits, out);
}
